// TimeMixing_44135083934193
// MI455X (gfx1250) — matrix operand order repaired and checked by reading, not compiled, not run on hardware
//
#include <hip/hip_runtime.h>
#include <hip/hip_bf16.h>

// ---------------------------------------------------------------------------
// RWKV-style TimeMixing for MI455X (gfx1250, wave32, WMMA).
// B=16384, D=DK=1024.  Two 16384x1024x1024 GEMMs via v_wmma_f32_16x16x32_bf16
// (f32 accumulate), full elementwise epilogue fused into GEMM1.
// ---------------------------------------------------------------------------

typedef __attribute__((ext_vector_type(16))) __bf16 v16bf;
typedef __attribute__((ext_vector_type(8)))  __bf16 v8bf;
typedef __attribute__((ext_vector_type(4)))  __bf16 v4bf;
typedef __attribute__((ext_vector_type(8)))  float  v8f;
typedef __attribute__((ext_vector_type(4)))  float  v4f;

#define B_ROWS 16384
#define D_DIM  1024
#define DK_DIM 1024

// ---------------------------------------------------------------------------
// Prep: mixed = mix*x + (1-mix)*x_prev, cast to bf16 (4 elements/thread).
// ---------------------------------------------------------------------------
__global__ __launch_bounds__(256)
void prep_mixed_kernel(const float* __restrict__ x,
                       const float* __restrict__ xp,
                       const float* __restrict__ mixp,
                       __bf16* __restrict__ out, int n4) {
  int i = blockIdx.x * blockDim.x + threadIdx.x;
  if (i >= n4) return;
  float mix = mixp[0];
  v4f a = ((const v4f*)x)[i];
  v4f b = ((const v4f*)xp)[i];
  v4bf o;
  #pragma unroll
  for (int e = 0; e < 4; ++e)
    o[e] = (__bf16)(mix * a[e] + (1.0f - mix) * b[e]);
  ((v4bf*)out)[i] = o;
}

// ---------------------------------------------------------------------------
// Cast a f32 weight matrix to bf16 (4 elements/thread).
// ---------------------------------------------------------------------------
__global__ __launch_bounds__(256)
void cast_bf16_kernel(const float* __restrict__ w, __bf16* __restrict__ out, int n4) {
  int i = blockIdx.x * blockDim.x + threadIdx.x;
  if (i >= n4) return;
  v4f a = ((const v4f*)w)[i];
  v4bf o;
  #pragma unroll
  for (int e = 0; e < 4; ++e) o[e] = (__bf16)a[e];
  ((v4bf*)out)[i] = o;
}

// ---------------------------------------------------------------------------
// Fragment loaders (row-major source, K-contiguous).
//
// A (16x32 bf16, M x K): lane l<16 holds row M=l, K 0-15; lane l>=16 holds
// row M=l-16, K 16-31.  -> one 32B contiguous load.
// B (32x16 bf16, K x N): lane l<16 holds col N=l, K 0-15; lane l>=16 holds
// col N=l-16, K 16-31.  -> one 32B contiguous load.
// ---------------------------------------------------------------------------
__device__ __forceinline__ v16bf load_frag_A(const __bf16* __restrict__ base,
                                             int row, int ld, int k0, int khalf) {
  const __bf16* p = base + (size_t)row * ld + k0 + khalf * 16;
  return *(const v16bf*)p;
}

__device__ __forceinline__ v16bf load_frag_B(const __bf16* __restrict__ base,
                                             int col, int ld, int k0, int khalf) {
  const __bf16* p = base + (size_t)col * ld + k0 + khalf * 16;
  return *(const v16bf*)p;
}

// ---------------------------------------------------------------------------
// GEMM1 + fused RWKV epilogue.
//   Kacc = mixed[B,D] @ Wk^T[D,DK] + bk        (Wk row n is K-contiguous)
//   e_bk = exp(bonus + K); wkv = (num + e_bk*K)/(den + e_bk)
//   rwkv = sigmoid(K)*wkv (bf16 -> ws);  num_new = w*num + e^K*K;
//   den_new = w*den + e^K;  w = exp(-exp(decay))
// 64x64 tile per wave, 8 waves/block.
// ---------------------------------------------------------------------------
__global__ __launch_bounds__(256)
void gemm1_rwkv_kernel(const __bf16* __restrict__ Abf,   // mixed [B, D]
                       const __bf16* __restrict__ Wkbf,  // [DK, D]
                       const float*  __restrict__ bk,
                       const float*  __restrict__ num,
                       const float*  __restrict__ den,
                       const float*  __restrict__ bonus,
                       const float*  __restrict__ decayp,
                       float* __restrict__ num_new,
                       float* __restrict__ den_new,
                       __bf16* __restrict__ rwkv) {
  const int lane  = threadIdx.x & 31;
  const int wave  = threadIdx.x >> 5;
  const int tile  = blockIdx.x * 8 + wave;
  const int ntn   = DK_DIM / 64;
  const int m0    = (tile / ntn) * 64;
  const int n0    = (tile % ntn) * 64;
  const int khalf = lane >> 4;
  const int lid   = lane & 15;

  v8f acc[4][4] = {};
  for (int k0 = 0; k0 < D_DIM; k0 += 32) {
    v16bf a[4], b[4];
    #pragma unroll
    for (int i = 0; i < 4; ++i)
      a[i] = load_frag_A(Abf, m0 + i * 16 + lid, D_DIM, k0, khalf);
    #pragma unroll
    for (int j = 0; j < 4; ++j)
      b[j] = load_frag_B(Wkbf, n0 + j * 16 + lid, D_DIM, k0, khalf);
    #pragma unroll
    for (int i = 0; i < 4; ++i)
      #pragma unroll
      for (int j = 0; j < 4; ++j)
        acc[i][j] = __builtin_amdgcn_wmma_f32_16x16x32_bf16(
            false, a[i], false, b[j], (short)0, acc[i][j], false, false);
  }

  const float w = __expf(-__expf(decayp[0]));
  #pragma unroll
  for (int i = 0; i < 4; ++i) {
    #pragma unroll
    for (int j = 0; j < 4; ++j) {
      const int n = n0 + j * 16 + lid;
      const float bkn = bk[n];
      const float bon = bonus[n];
      #pragma unroll
      for (int r = 0; r < 8; ++r) {
        const int m = m0 + i * 16 + r + khalf * 8;   // C layout: v_r -> M=r / r+8
        const size_t idx = (size_t)m * DK_DIM + n;
        const float kv  = acc[i][j][r] + bkn;
        const float ebk = __expf(bon + kv);
        const float nm  = num[idx];
        const float dn  = den[idx];
        const float wkv = (nm + ebk * kv) / (dn + ebk);
        const float sig = 1.0f / (1.0f + __expf(-kv));
        const float ek  = __expf(kv);
        rwkv[idx]    = (__bf16)(sig * wkv);
        num_new[idx] = w * nm + ek * kv;
        den_new[idx] = w * dn + ek;
      }
    }
  }
}

// ---------------------------------------------------------------------------
// GEMM2: out = rwkv[B,DK] @ Wf^T[DK,D] + bf   (Wf row n is K-contiguous)
// ---------------------------------------------------------------------------
__global__ __launch_bounds__(256)
void gemm2_kernel(const __bf16* __restrict__ Abf,   // rwkv [B, DK]
                  const __bf16* __restrict__ Wfbf,  // [D, DK]
                  const float*  __restrict__ bf_,
                  float* __restrict__ out) {
  const int lane  = threadIdx.x & 31;
  const int wave  = threadIdx.x >> 5;
  const int tile  = blockIdx.x * 8 + wave;
  const int ntn   = D_DIM / 64;
  const int m0    = (tile / ntn) * 64;
  const int n0    = (tile % ntn) * 64;
  const int khalf = lane >> 4;
  const int lid   = lane & 15;

  v8f acc[4][4] = {};
  for (int k0 = 0; k0 < DK_DIM; k0 += 32) {
    v16bf a[4], b[4];
    #pragma unroll
    for (int i = 0; i < 4; ++i)
      a[i] = load_frag_A(Abf, m0 + i * 16 + lid, DK_DIM, k0, khalf);
    #pragma unroll
    for (int j = 0; j < 4; ++j)
      b[j] = load_frag_B(Wfbf, n0 + j * 16 + lid, DK_DIM, k0, khalf);
    #pragma unroll
    for (int i = 0; i < 4; ++i)
      #pragma unroll
      for (int j = 0; j < 4; ++j)
        acc[i][j] = __builtin_amdgcn_wmma_f32_16x16x32_bf16(
            false, a[i], false, b[j], (short)0, acc[i][j], false, false);
  }

  #pragma unroll
  for (int i = 0; i < 4; ++i) {
    #pragma unroll
    for (int j = 0; j < 4; ++j) {
      const int n = n0 + j * 16 + lid;
      const float bias = bf_[n];
      #pragma unroll
      for (int r = 0; r < 8; ++r) {
        const int m = m0 + i * 16 + r + khalf * 8;
        out[(size_t)m * D_DIM + n] = acc[i][j][r] + bias;
      }
    }
  }
}

// ---------------------------------------------------------------------------
// Launch.  Inputs: x, x_prev, num, den, Wk, bk, mix_k, Wf, bf, bonus, decay.
// d_out = [ out(B*D) | num_new(B*DK) | den_new(B*DK) ]  (f32).
// Workspace: mixed_bf16(32MB) | rwkv_bf16(32MB) | Wk_bf16(2MB) | Wf_bf16(2MB).
// ---------------------------------------------------------------------------
extern "C" void kernel_launch(void* const* d_in, const int* in_sizes, int n_in,
                              void* d_out, int out_size, void* d_ws, size_t ws_size,
                              hipStream_t stream) {
  const float* x      = (const float*)d_in[0];
  const float* x_prev = (const float*)d_in[1];
  const float* num    = (const float*)d_in[2];
  const float* den    = (const float*)d_in[3];
  const float* Wk     = (const float*)d_in[4];
  const float* bk     = (const float*)d_in[5];
  const float* mix_k  = (const float*)d_in[6];
  const float* Wf     = (const float*)d_in[7];
  const float* bfv    = (const float*)d_in[8];
  const float* bonus  = (const float*)d_in[9];
  const float* decay  = (const float*)d_in[10];

  float* out     = (float*)d_out;
  float* num_new = out + (size_t)B_ROWS * D_DIM;
  float* den_new = num_new + (size_t)B_ROWS * DK_DIM;

  char* ws = (char*)d_ws;
  __bf16* mixed_bf = (__bf16*)(ws);
  __bf16* rwkv_bf  = (__bf16*)(ws + (size_t)B_ROWS * D_DIM * 2);
  __bf16* Wk_bf    = (__bf16*)(ws + (size_t)B_ROWS * D_DIM * 4);
  __bf16* Wf_bf    = (__bf16*)(ws + (size_t)B_ROWS * D_DIM * 4 + (size_t)DK_DIM * D_DIM * 2);

  // 1) mix + cast activations
  {
    int n4 = B_ROWS * D_DIM / 4;
    prep_mixed_kernel<<<n4 / 256, 256, 0, stream>>>(x, x_prev, mix_k, mixed_bf, n4);
  }
  // 2) cast weights
  {
    int w4 = DK_DIM * D_DIM / 4;
    cast_bf16_kernel<<<w4 / 256, 256, 0, stream>>>(Wk, Wk_bf, w4);
    cast_bf16_kernel<<<w4 / 256, 256, 0, stream>>>(Wf, Wf_bf, w4);
  }
  // 3) GEMM1 + fused RWKV epilogue
  {
    int tiles = (B_ROWS / 64) * (DK_DIM / 64);   // 4096
    gemm1_rwkv_kernel<<<tiles / 8, 256, 0, stream>>>(
        mixed_bf, Wk_bf, bk, num, den, bonus, decay, num_new, den_new, rwkv_bf);
  }
  // 4) GEMM2 + bias
  {
    int tiles = (B_ROWS / 64) * (D_DIM / 64);    // 4096
    gemm2_kernel<<<tiles / 8, 256, 0, stream>>>(rwkv_bf, Wf_bf, bfv, out);
  }
}
